// Module_27221502722464
// MI455X (gfx1250) — matrix operand order repaired and checked by reading, not compiled, not run on hardware
//
#include <hip/hip_runtime.h>
#include <hip/hip_bf16.h>

// ---------------------------------------------------------------------------
// Problem constants (from reference)
// ---------------------------------------------------------------------------
#define N_DIM   128
#define N_QUERY 2048
#define N_KEY   512
#define M_TOTAL (N_QUERY * N_KEY)        // 1,048,576 (q,k) rows
#define LN_EPS  1e-5f

typedef __attribute__((ext_vector_type(16))) __bf16 v16bf;
typedef __attribute__((ext_vector_type(8)))  float  v8f;

union AFrag { v16bf v; __bf16 e[16]; };

// ---------------------------------------------------------------------------
// Kernel 0: per-query dots  qmu[q] = Q[q]·Wmu[:128]+b_mu,  qlv likewise
// ---------------------------------------------------------------------------
__global__ void qdots_kernel(const float* __restrict__ Q,
                             const float* __restrict__ Wmu, const float* __restrict__ bmu,
                             const float* __restrict__ Wlv, const float* __restrict__ blv,
                             float* __restrict__ qmu, float* __restrict__ qlv) {
    int q = blockIdx.x * blockDim.x + threadIdx.x;
    if (q >= N_QUERY) return;
    const float* r = Q + (size_t)q * N_DIM;
    float a = 0.f, b = 0.f;
    #pragma unroll 8
    for (int i = 0; i < N_DIM; ++i) {
        float x = r[i];
        a = fmaf(x, Wmu[i], a);
        b = fmaf(x, Wlv[i], b);
    }
    qmu[q] = a + bmu[0];
    qlv[q] = b + blv[0];
}

// ---------------------------------------------------------------------------
// Kernel 1: fused WMMA GEMM over K-rows.
//  Per wave: 16 rows of K_flat[M,128].
//  B = [ W_pr1^T (128x128) | Wmu[d:], Wlv[d:] as cols 0,1 of a 9th tile ]
//  D = A x B  via v_wmma_f32_16x16x32_bf16, 9 tiles x 4 k-steps.
//  Epilogue: relu+W_pr2 dot -> mu_prior; sigma/mu_post/KL/samples.
// ---------------------------------------------------------------------------
__global__ void __launch_bounds__(256)
prior_wmma_kernel(const float* __restrict__ Kp,   const float* __restrict__ eps,
                  const float* __restrict__ Wpr1, const float* __restrict__ bpr1,
                  const float* __restrict__ Wpr2, const float* __restrict__ bpr2,
                  const float* __restrict__ Wmu,  const float* __restrict__ Wlv,
                  const float* __restrict__ qmu,  const float* __restrict__ qlv,
                  float* __restrict__ samples,    float* __restrict__ klpart) {
    // 9 n-tiles * 4 k-steps, each fragment = 32 lanes * 16 bf16 = 512 elems
    __shared__ __bf16 Bfrag[36 * 512];             // 36 KB
    __shared__ float  klw[8];

    const int tid  = threadIdx.x;
    const int wave = tid >> 5;
    const int lane = tid & 31;

    // ---- build B fragments in LDS in contiguous-k B-layout (16-bit, 32x16):
    //  lane l<16: col n=l,    k = i        ;  lane l>=16: col n=l-16, k = 16+i
    for (int idx = tid; idx < 36 * 32; idx += 256) {
        int frag = idx >> 5, bl = idx & 31;
        int t = frag >> 2, s = frag & 3;
        int kbase = s * 32 + ((bl & 16) ? 16 : 0);
        int n = bl & 15;
        __bf16* dst = &Bfrag[frag * 512 + bl * 16];
        if (t < 8) {
            int col = t * 16 + n;                 // output feature e
            #pragma unroll
            for (int i = 0; i < 16; ++i)
                dst[i] = (__bf16)Wpr1[(size_t)col * N_DIM + kbase + i];
        } else {                                  // 9th tile: Wmu/Wlv K-halves
            #pragma unroll
            for (int i = 0; i < 16; ++i) {
                float v = 0.f;
                if (n == 0)      v = Wmu[N_DIM + kbase + i];
                else if (n == 1) v = Wlv[N_DIM + kbase + i];
                dst[i] = (__bf16)v;
            }
        }
    }
    __syncthreads();

    // ---- load A fragments (16 rows x 128 f32 -> bf16), contiguous-k A-layout 16x32:
    //  lane<16: k in 32s+0..15; lane>=16: k in 32s+16..31
    const long row_m = ((long)blockIdx.x * 8 + wave) * 16 + (lane & 15);
    const float* arow = Kp + row_m * N_DIM;
    AFrag a[4];
    #pragma unroll
    for (int s = 0; s < 4; ++s) {
        int base = s * 32 + ((lane & 16) ? 16 : 0);
        float4 c0a = *(const float4*)(arow + base);
        float4 c0b = *(const float4*)(arow + base + 4);
        float4 c1a = *(const float4*)(arow + base + 8);
        float4 c1b = *(const float4*)(arow + base + 12);
        const float f0[8] = {c0a.x, c0a.y, c0a.z, c0a.w, c0b.x, c0b.y, c0b.z, c0b.w};
        const float f1[8] = {c1a.x, c1a.y, c1a.z, c1a.w, c1b.x, c1b.y, c1b.z, c1b.w};
        #pragma unroll
        for (int i = 0; i < 8; ++i) {
            a[s].e[i]     = (__bf16)f0[i];
            a[s].e[8 + i] = (__bf16)f1[i];
        }
    }

    // ---- GEMM: 36 x v_wmma_f32_16x16x32_bf16 per wave
    v8f acc[9] = {};
    #pragma unroll
    for (int t = 0; t < 9; ++t) {
        #pragma unroll
        for (int s = 0; s < 4; ++s) {
            v16bf b = *(const v16bf*)&Bfrag[(t * 4 + s) * 512 + lane * 16];
            acc[t] = __builtin_amdgcn_wmma_f32_16x16x32_bf16(
                false, a[s].v, false, b, (short)0, acc[t], false, false);
        }
    }

    // ---- epilogue: relu + dot with W_pr2 (per-row, cross-lane over 128 cols)
    float hsum[8] = {0.f, 0.f, 0.f, 0.f, 0.f, 0.f, 0.f, 0.f};
    const int nb = lane & 15;
    #pragma unroll
    for (int t = 0; t < 8; ++t) {
        int col = t * 16 + nb;
        float bias = bpr1[col];
        float w2   = Wpr2[col];
        #pragma unroll
        for (int r = 0; r < 8; ++r) {
            float h = fmaxf(acc[t][r] + bias, 0.f);
            hsum[r] = fmaf(h, w2, hsum[r]);
        }
    }
    #pragma unroll
    for (int r = 0; r < 8; ++r)
        #pragma unroll
        for (int off = 1; off < 16; off <<= 1)
            hsum[r] += __shfl_xor(hsum[r], off, 32);

    // dmu / dlv live in cols 0/1 of tile 8 (lanes 0/1 and 16/17)
    const int srcbase = lane & 16;
    float dmu[8], dlv[8];
    #pragma unroll
    for (int r = 0; r < 8; ++r) {
        dmu[r] = __shfl(acc[8][r], srcbase | 0, 32);
        dlv[r] = __shfl(acc[8][r], srcbase | 1, 32);
    }

    // ---- per-(q,k) scalar math on 16 lanes (rows 0..7 -> lanes 0..7,
    //      rows 8..15 -> lanes 16..23)
    float klv = 0.f;
    if ((lane & 15) < 8) {
        int   r    = lane & 7;
        long  grow = ((long)blockIdx.x * 8 + wave) * 16 + ((lane & 16) ? 8 : 0) + r;
        int   q    = (int)(grow >> 9);            // / N_KEY
        float mup0 = qmu[q] + dmu[r];
        float lv   = qlv[q] + dlv[r];
        float s2   = expf(lv);                    // sigma_post^2
        float sig  = expf(0.5f * lv);
        float mu   = mup0 - 0.5f * s2;
        float mupr = hsum[r] + bpr2[0] - 0.5f;    // - SIGMA_PRIOR^2/2
        float diff = mu - mupr;
        klv = -0.5f * lv + 0.5f * (s2 + diff * diff) - 0.5f;
        float x = mu + sig * eps[grow];
        samples[grow] = expf(x);                  // TEMP == 1
    }
    #pragma unroll
    for (int off = 16; off; off >>= 1) klv += __shfl_xor(klv, off, 32);
    if (lane == 0) klw[wave] = klv;
    __syncthreads();
    if (tid == 0) {
        float s = 0.f;
        #pragma unroll
        for (int i = 0; i < 8; ++i) s += klw[i];
        klpart[blockIdx.x] = s;
    }
}

// ---------------------------------------------------------------------------
// Kernel 2: per-query softmax over 512 samples, weights @ V, LayerNorm.
// One workgroup (256 threads) per query; V streamed once with b128 loads.
// ---------------------------------------------------------------------------
__global__ void __launch_bounds__(256)
ctx_kernel(const float* __restrict__ V, const float* __restrict__ samples,
           const float* __restrict__ gamma, const float* __restrict__ beta,
           float* __restrict__ out) {
    __shared__ float sw[N_KEY];
    __shared__ float red[16];
    __shared__ float part[8][N_DIM];
    __shared__ float stats[2];

    const int q = blockIdx.x, tid = threadIdx.x;
    const int wv = tid >> 5, ln = tid & 31;
    const float* srow = samples + (size_t)q * N_KEY;

    float s0 = srow[tid], s1 = srow[tid + 256];
    // ---- softmax max
    float m = fmaxf(s0, s1);
    for (int off = 16; off; off >>= 1) m = fmaxf(m, __shfl_xor(m, off, 32));
    if (ln == 0) red[wv] = m;
    __syncthreads();
    if (tid == 0) {
        float mm = red[0];
        for (int i = 1; i < 8; ++i) mm = fmaxf(mm, red[i]);
        stats[0] = mm;
    }
    __syncthreads();
    float mx = stats[0];
    float e0 = expf(s0 - mx), e1 = expf(s1 - mx);
    float sum = e0 + e1;
    for (int off = 16; off; off >>= 1) sum += __shfl_xor(sum, off, 32);
    if (ln == 0) red[wv] = sum;
    __syncthreads();
    if (tid == 0) {
        float t = 0.f;
        for (int i = 0; i < 8; ++i) t += red[i];
        stats[1] = t;
    }
    __syncthreads();
    float inv = 1.f / stats[1];
    sw[tid] = e0 * inv;
    sw[tid + 256] = e1 * inv;
    __syncthreads();

    // ---- context[d] = sum_k w_k * V[q,k,d]
    // lane covers 4 d's (float4); 8 wave-groups stride over k -> b128 loads
    const int g  = tid >> 5;          // 0..7  (k-stride group)
    const int dv = (tid & 31) * 4;    // 0..124
    const float* vb = V + (size_t)q * N_KEY * N_DIM + dv;
    float4 acc = make_float4(0.f, 0.f, 0.f, 0.f);
    #pragma unroll 4
    for (int k = g; k < N_KEY; k += 8) {
        float  w = sw[k];
        float4 x = *(const float4*)(vb + (size_t)k * N_DIM);
        acc.x = fmaf(w, x.x, acc.x);
        acc.y = fmaf(w, x.y, acc.y);
        acc.z = fmaf(w, x.z, acc.z);
        acc.w = fmaf(w, x.w, acc.w);
    }
    *(float4*)&part[g][dv] = acc;
    __syncthreads();

    // ---- combine 8 partials, LayerNorm over 128
    float c = 0.f;
    if (tid < N_DIM) {
        c = part[0][tid];
        #pragma unroll
        for (int i = 1; i < 8; ++i) c += part[i][tid];
    }
    float ssum = c, ssq = c * c;
    for (int off = 16; off; off >>= 1) {
        ssum += __shfl_xor(ssum, off, 32);
        ssq  += __shfl_xor(ssq,  off, 32);
    }
    if (tid < N_DIM && (tid & 31) == 0) { red[tid >> 5] = ssum; red[8 + (tid >> 5)] = ssq; }
    __syncthreads();
    if (tid == 0) {
        float s = 0.f, s2 = 0.f;
        for (int i = 0; i < 4; ++i) { s += red[i]; s2 += red[8 + i]; }
        float mean = s * (1.f / N_DIM);
        float var  = s2 * (1.f / N_DIM) - mean * mean;
        stats[0] = mean;
        stats[1] = rsqrtf(var + LN_EPS);
    }
    __syncthreads();
    if (tid < N_DIM) {
        float o = (c - stats[0]) * stats[1] * gamma[tid] + beta[tid];
        out[(size_t)q * N_DIM + tid] = o;
    }
}

// ---------------------------------------------------------------------------
// Kernel 3: deterministic KL partial reduction -> mean
// ---------------------------------------------------------------------------
__global__ void kl_reduce_kernel(const float* __restrict__ parts, float* __restrict__ out) {
    __shared__ float red[8];
    float s = 0.f;
    for (int i = threadIdx.x; i < 8192; i += 256) s += parts[i];
    for (int off = 16; off; off >>= 1) s += __shfl_xor(s, off, 32);
    if ((threadIdx.x & 31) == 0) red[threadIdx.x >> 5] = s;
    __syncthreads();
    if (threadIdx.x == 0) {
        float t = 0.f;
        for (int i = 0; i < 8; ++i) t += red[i];
        out[0] = t / (float)M_TOTAL;
    }
}

// ---------------------------------------------------------------------------
extern "C" void kernel_launch(void* const* d_in, const int* in_sizes, int n_in,
                              void* d_out, int out_size, void* d_ws, size_t ws_size,
                              hipStream_t stream) {
    const float* Q    = (const float*)d_in[0];
    const float* K    = (const float*)d_in[1];
    const float* V    = (const float*)d_in[2];
    const float* eps  = (const float*)d_in[3];
    const float* Wmu  = (const float*)d_in[4];
    const float* bmu  = (const float*)d_in[5];
    const float* Wlv  = (const float*)d_in[6];
    const float* blv  = (const float*)d_in[7];
    const float* Wpr1 = (const float*)d_in[8];
    const float* bpr1 = (const float*)d_in[9];
    const float* Wpr2 = (const float*)d_in[10];
    const float* bpr2 = (const float*)d_in[11];
    const float* gam  = (const float*)d_in[12];
    const float* bet  = (const float*)d_in[13];

    float* out = (float*)d_out;                 // [context(2048*128) | kl(1)]
    float* ws  = (float*)d_ws;
    float* qmu     = ws;                        // 2048
    float* qlv     = ws + 2048;                 // 2048
    float* samples = ws + 4096;                 // 1,048,576
    float* klpart  = ws + 4096 + M_TOTAL;       // 8192

    qdots_kernel<<<N_QUERY / 256, 256, 0, stream>>>(Q, Wmu, bmu, Wlv, blv, qmu, qlv);
    prior_wmma_kernel<<<M_TOTAL / 128, 256, 0, stream>>>(
        K, eps, Wpr1, bpr1, Wpr2, bpr2, Wmu, Wlv, qmu, qlv, samples, klpart);
    ctx_kernel<<<N_QUERY, 256, 0, stream>>>(V, samples, gam, bet, out);
    kl_reduce_kernel<<<1, 256, 0, stream>>>(klpart, out + (size_t)N_QUERY * N_DIM);
}
